// RACP_64312840290535
// MI455X (gfx1250) — compile-verified, matrix operand order checked by reading, not run on hardware
//
#include <hip/hip_runtime.h>
#include <cstdint>
#include <cstddef>

#define BATCH 2048
#define NTOK  25
#define DMODEL 768
#define NHEAD 8
#define DHEAD 96

typedef __attribute__((ext_vector_type(16))) __bf16 v16bf;
typedef __attribute__((ext_vector_type(8)))  __bf16 v8bf;
typedef __attribute__((ext_vector_type(8)))  float  v8f;

// ---------- helpers ----------
__device__ __forceinline__ unsigned short f2bf(float f) {
  union { float f; unsigned u; } x; x.f = f;
  unsigned u = x.u;
  unsigned r = u + 0x7FFFu + ((u >> 16) & 1u);   // RNE
  return (unsigned short)(r >> 16);
}
__device__ __forceinline__ float bf2f(unsigned short b) {
  union { unsigned u; float f; } x; x.u = ((unsigned)b) << 16; return x.f;
}
__device__ __forceinline__ float wave_sum(float v) {
#pragma unroll
  for (int m = 16; m > 0; m >>= 1) v += __shfl_xor(v, m, 32);
  return v;
}

// ---------- weight convert: W[K,N] f32 -> Bt[N,K] bf16 ----------
__global__ void k_transpose_bf16(const float* __restrict__ W, unsigned short* __restrict__ Bt,
                                 int K, int N) {
  int i = blockIdx.x * blockDim.x + threadIdx.x;
  if (i >= K * N) return;
  int n = i / K, k = i % K;
  Bt[(size_t)n * K + k] = f2bf(W[(size_t)k * N + n]);
}

// ---------- WMMA fragment load (16-bit A/B layout, 16x32 MxK per ISA) ----------
__device__ __forceinline__ v16bf load_frag(const unsigned short* __restrict__ base,
                                           int row16, int stride, int k0, int lane) {
  int lr = lane & 15, hi = lane >> 4;
  const v8bf* p = (const v8bf*)(base + (size_t)(row16 + lr) * stride + k0 + hi * 8);
  v8bf lo = p[0];
  v8bf hh = p[2];   // +16 elements
  v16bf r;
#pragma unroll
  for (int i = 0; i < 8; ++i) { r[i] = lo[i]; r[i + 8] = hh[i]; }
  return r;
}

__device__ __forceinline__ void store_tile(const v8f& acc, int m0, int n0,
                                           const float* __restrict__ bias,
                                           float* __restrict__ Cf,
                                           unsigned short* __restrict__ Cb,
                                           int N, int relu, int lane) {
  int lr = lane & 15, hi = lane >> 4;
  int col = n0 + lr;
  float bv = bias ? bias[col] : 0.f;
#pragma unroll
  for (int i = 0; i < 8; ++i) {
    int row = m0 + hi * 8 + i;
    float v = acc[i] + bv;
    if (relu) v = v > 0.f ? v : 0.f;
    size_t off = (size_t)row * N + col;
    if (Cf) Cf[off] = v;
    if (Cb) Cb[off] = f2bf(v);
  }
}

// ---------- generic bf16 WMMA GEMM: C = A[M,K] * Bt[N,K]^T (+bias)(+relu) ----------
__global__ __launch_bounds__(256)
void k_gemm_wmma(const unsigned short* __restrict__ A,
                 const unsigned short* __restrict__ Bt,
                 const float* __restrict__ bias,
                 float* __restrict__ Cf, unsigned short* __restrict__ Cb,
                 int M, int N, int K, int relu) {
  int lane = threadIdx.x & 31;
  int wave = (int)((blockIdx.x * blockDim.x + threadIdx.x) >> 5);
  int ntn = N >> 5;
  int total = (M >> 5) * ntn;
  if (wave >= total) return;           // uniform per wave
  int tm = wave / ntn, tn = wave % ntn;
  int m0 = tm << 5, n0 = tn << 5;
  v8f acc00 = {}, acc01 = {}, acc10 = {}, acc11 = {};
  for (int k = 0; k < K; k += 32) {
    v16bf a0 = load_frag(A,  m0,      K, k, lane);
    v16bf a1 = load_frag(A,  m0 + 16, K, k, lane);
    v16bf b0 = load_frag(Bt, n0,      K, k, lane);
    v16bf b1 = load_frag(Bt, n0 + 16, K, k, lane);
    acc00 = __builtin_amdgcn_wmma_f32_16x16x32_bf16(false, a0, false, b0, (short)0, acc00, false, false);
    acc01 = __builtin_amdgcn_wmma_f32_16x16x32_bf16(false, a0, false, b1, (short)0, acc01, false, false);
    acc10 = __builtin_amdgcn_wmma_f32_16x16x32_bf16(false, a1, false, b0, (short)0, acc10, false, false);
    acc11 = __builtin_amdgcn_wmma_f32_16x16x32_bf16(false, a1, false, b1, (short)0, acc11, false, false);
  }
  store_tile(acc00, m0,      n0,      bias, Cf, Cb, N, relu, lane);
  store_tile(acc01, m0,      n0 + 16, bias, Cf, Cb, N, relu, lane);
  store_tile(acc10, m0 + 16, n0,      bias, Cf, Cb, N, relu, lane);
  store_tile(acc11, m0 + 16, n0 + 16, bias, Cf, Cb, N, relu, lane);
}

// ---------- embedding: page-ad tokens [B*25, 768] bf16 + NaN mask ----------
__global__ void k_tokens(const float* __restrict__ ad, const float* __restrict__ loc,
                         const float* __restrict__ cat, const float* __restrict__ sq,
                         const float* __restrict__ sp, const float* __restrict__ at,
                         const float* __restrict__ ap, const float* __restrict__ ck,
                         const int* __restrict__ feat, unsigned short* __restrict__ tok,
                         int* __restrict__ mask) {
  int lane = threadIdx.x & 31;
  int gw = (int)((blockIdx.x * blockDim.x + threadIdx.x) >> 5);
  if (gw >= BATCH * NTOK) return;
  int b = gw / NTOK, t = gw % NTOK, page = t / 5, adix = t % 5;
  const int* f = feat + (size_t)b * 875 + 40 + page * 167 + adix * 33;
  int click = feat[(size_t)b * 875 + 40 + page * 167 + 166];
  unsigned short* o = tok + (size_t)gw * DMODEL;
#pragma unroll
  for (int j = 0; j < 3; ++j) o[j*32+lane] = f2bf(ad[((size_t)j*10000 + f[j])*32 + lane]);
  o[3*32+lane] = f2bf(ad[((size_t)3*10000 + f[15])*32 + lane]);
#pragma unroll
  for (int j = 0; j < 6; ++j) o[(4+j)*32+lane] = f2bf(ad[((size_t)(4+j)*10000 + f[27+j])*32 + lane]);
#pragma unroll
  for (int j = 0; j < 3; ++j) o[(10+j)*32+lane] = f2bf(loc[((size_t)j*1000 + f[3+j])*32 + lane]);
#pragma unroll
  for (int j = 0; j < 3; ++j) o[(13+j)*32+lane] = f2bf(cat[((size_t)j*5000 + f[6+j])*32 + lane]);
  o[16*32+lane] = f2bf(sq[(size_t)f[9]*32 + lane]);
  float s = 0.f;
#pragma unroll
  for (int j = 0; j < 5; ++j) s += sp[(size_t)f[10+j]*32 + lane];
  o[17*32+lane] = f2bf(s);
#pragma unroll
  for (int j = 0; j < 3; ++j) o[(18+j)*32+lane] = f2bf(cat[((size_t)j*5000 + f[16+j])*32 + lane]);
  s = 0.f;
#pragma unroll
  for (int j = 0; j < 4; ++j) s += at[(size_t)f[19+j]*32 + lane];
  o[21*32+lane] = f2bf(s);
  s = 0.f;
#pragma unroll
  for (int j = 0; j < 4; ++j) s += ap[(size_t)f[23+j]*32 + lane];
  o[22*32+lane] = f2bf(s);
  o[23*32+lane] = f2bf(ck[(size_t)click*32 + lane]);
  if (lane == 0) mask[gw] = (f[32] == 2) ? 1 : 0;
}

// ---------- embedding: target [B,704] (f32+bf16) and user [B,256] ----------
__global__ void k_target_user(const float* __restrict__ ue, const float* __restrict__ ad,
                              const float* __restrict__ loc, const float* __restrict__ cat,
                              const float* __restrict__ sq, const float* __restrict__ sp,
                              const float* __restrict__ at, const float* __restrict__ ap,
                              const int* __restrict__ feat, float* __restrict__ tf,
                              unsigned short* __restrict__ tb, float* __restrict__ uf_out) {
  int lane = threadIdx.x & 31;
  int b = (int)((blockIdx.x * blockDim.x + threadIdx.x) >> 5);
  if (b >= BATCH) return;
  const int* f = feat + (size_t)b * 875 + 8;
  float* T = tf + (size_t)b * 704;
  unsigned short* Tb = tb + (size_t)b * 704;
  auto emit = [&](int c, float v) { T[c*32+lane] = v; Tb[c*32+lane] = f2bf(v); };
#pragma unroll
  for (int j = 0; j < 3; ++j) emit(j, ad[((size_t)j*10000 + f[j])*32 + lane]);
  emit(3, ad[((size_t)3*10000 + f[15])*32 + lane]);
#pragma unroll
  for (int j = 0; j < 5; ++j) emit(4+j, ad[((size_t)(4+j)*10000 + f[27+j])*32 + lane]);
#pragma unroll
  for (int j = 0; j < 3; ++j) emit(9+j, loc[((size_t)j*1000 + f[3+j])*32 + lane]);
#pragma unroll
  for (int j = 0; j < 3; ++j) emit(12+j, cat[((size_t)j*5000 + f[6+j])*32 + lane]);
  emit(15, sq[(size_t)f[9]*32 + lane]);
  float s = 0.f;
#pragma unroll
  for (int j = 0; j < 5; ++j) s += sp[(size_t)f[10+j]*32 + lane];
  emit(16, s);
#pragma unroll
  for (int j = 0; j < 3; ++j) emit(17+j, cat[((size_t)j*5000 + f[16+j])*32 + lane]);
  s = 0.f;
#pragma unroll
  for (int j = 0; j < 4; ++j) s += at[(size_t)f[19+j]*32 + lane];
  emit(20, s);
  s = 0.f;
#pragma unroll
  for (int j = 0; j < 4; ++j) s += ap[(size_t)f[23+j]*32 + lane];
  emit(21, s);
  const int* u = feat + (size_t)b * 875;
#pragma unroll
  for (int j = 0; j < 8; ++j)
    uf_out[(size_t)b*256 + j*32 + lane] = ue[((size_t)j*1000 + u[j])*32 + lane];
}

// ---------- query-seq mean pool [B,384] ----------
__global__ void k_qpool(const float* __restrict__ ad, const float* __restrict__ loc,
                        const float* __restrict__ cat, const float* __restrict__ sq,
                        const float* __restrict__ sp, const float* __restrict__ ck,
                        const int* __restrict__ feat, float* __restrict__ qp) {
  int lane = threadIdx.x & 31;
  int b = (int)((blockIdx.x * blockDim.x + threadIdx.x) >> 5);
  if (b >= BATCH) return;
  float a[12];
#pragma unroll
  for (int c = 0; c < 12; ++c) a[c] = 0.f;
  for (int page = 0; page < 5; ++page) {
    const int* f = feat + (size_t)b * 875 + 40 + page * 167;
    int click = f[166];
#pragma unroll
    for (int j = 0; j < 3; ++j) a[j]   += ad[((size_t)j*10000 + f[j])*32 + lane];
#pragma unroll
    for (int j = 0; j < 3; ++j) a[3+j] += loc[((size_t)j*1000 + f[3+j])*32 + lane];
#pragma unroll
    for (int j = 0; j < 3; ++j) a[6+j] += cat[((size_t)j*5000 + f[6+j])*32 + lane];
    a[9] += sq[(size_t)f[9]*32 + lane];
    float s = 0.f;
#pragma unroll
    for (int j = 0; j < 5; ++j) s += sp[(size_t)f[10+j]*32 + lane];
    a[10] += s;
    a[11] += ck[(size_t)click*32 + lane];
  }
#pragma unroll
  for (int c = 0; c < 12; ++c) qp[(size_t)b*384 + c*32 + lane] = a[c] * 0.2f;
}

// ---------- AGSOM codebook + tiny QKV attention over 3 nodes ----------
__global__ void k_agsom(const float* __restrict__ item, const float* __restrict__ nodes,
                        const float* __restrict__ wq, const float* __restrict__ wk,
                        const float* __restrict__ wv, unsigned short* __restrict__ fin) {
  int lane = threadIdx.x & 31;
  int b = (int)((blockIdx.x * blockDim.x + threadIdx.x) >> 5);
  if (b >= BATCH) return;
  const float* it = item + (size_t)b * 64;
  float i0 = it[lane], i1 = it[lane + 32];
  float d2[3];
#pragma unroll
  for (int j = 0; j < 3; ++j) {
    float dx0 = i0 - nodes[j*64 + lane], dx1 = i1 - nodes[j*64 + lane + 32];
    d2[j] = wave_sum(dx0*dx0 + dx1*dx1);
  }
  int bmu = 0;
  if (d2[1] < d2[bmu]) bmu = 1;
  if (d2[2] < d2[bmu]) bmu = 2;
  float n0 = nodes[bmu*64 + lane], n1 = nodes[bmu*64 + lane + 32];
  float q0 = 0.f, q1 = 0.f;
  for (int k = 0; k < 64; ++k) {
    float iv = it[k];
    q0 += iv * wq[k*64 + lane];
    q1 += iv * wq[k*64 + lane + 32];
  }
  float kj0[3], kj1[3], vj0[3], vj1[3];
#pragma unroll
  for (int j = 0; j < 3; ++j) {
    float k0 = 0.f, k1 = 0.f, v0 = 0.f, v1 = 0.f;
    for (int k = 0; k < 64; ++k) {
      float nv = nodes[j*64 + k];
      k0 += nv * wk[k*64 + lane];  k1 += nv * wk[k*64 + lane + 32];
      v0 += nv * wv[k*64 + lane];  v1 += nv * wv[k*64 + lane + 32];
    }
    kj0[j] = k0; kj1[j] = k1; vj0[j] = v0; vj1[j] = v1;
  }
  float s[3];
#pragma unroll
  for (int j = 0; j < 3; ++j) s[j] = wave_sum(q0*kj0[j] + q1*kj1[j]) * 0.125f;
  float mx = fmaxf(s[0], fmaxf(s[1], s[2]));
  float e0 = __expf(s[0]-mx), e1 = __expf(s[1]-mx), e2 = __expf(s[2]-mx);
  float inv = 1.f / (e0 + e1 + e2);
  float w0 = e0*inv, w1 = e1*inv, w2 = e2*inv;
  float a0 = w0*vj0[0] + w1*vj0[1] + w2*vj0[2];
  float a1 = w0*vj1[0] + w1*vj1[1] + w2*vj1[2];
  unsigned short* o = fin + (size_t)b * 256;
  o[lane]        = f2bf(i0);      o[lane + 32]       = f2bf(i1);
  o[64 + lane]   = f2bf(n0);      o[64 + lane + 32]  = f2bf(n1);
  o[128 + lane]  = f2bf(a0);      o[128 + lane + 32] = f2bf(a1);
  o[192 + lane]  = f2bf(i0*n0);   o[192 + lane + 32] = f2bf(i1*n1);
}

// ---------- MHA softmax pooling over 25 tokens (wave per (b,h)) ----------
__global__ void k_attn(const float* __restrict__ q, const unsigned short* __restrict__ Kb,
                       const unsigned short* __restrict__ Vb, const int* __restrict__ mask,
                       unsigned short* __restrict__ pooled) {
  int lane = threadIdx.x & 31;
  int wave = (int)((blockIdx.x * blockDim.x + threadIdx.x) >> 5);
  if (wave >= BATCH * NHEAD) return;
  int b = wave / NHEAD, h = wave % NHEAD;
  const float* qr = q + (size_t)b * DMODEL + h * DHEAD;
  float q0 = qr[lane], q1 = qr[lane + 32], q2 = qr[lane + 64];
  const float scale = 0.10206207261596575f;   // 1/sqrt(96)
  float sc[NTOK];
#pragma unroll
  for (int j = 0; j < NTOK; ++j) {
    const unsigned short* kr = Kb + ((size_t)(b*NTOK + j)) * DMODEL + h * DHEAD;
    float p = q0*bf2f(kr[lane]) + q1*bf2f(kr[lane+32]) + q2*bf2f(kr[lane+64]);
    p = wave_sum(p) * scale;
    sc[j] = mask[b*NTOK + j] ? -1e9f : p;
  }
  float mx = sc[0];
#pragma unroll
  for (int j = 1; j < NTOK; ++j) mx = fmaxf(mx, sc[j]);
  float s = 0.f;
#pragma unroll
  for (int j = 0; j < NTOK; ++j) { sc[j] = __expf(sc[j] - mx); s += sc[j]; }
  float inv = 1.f / s;
  float a0 = 0.f, a1 = 0.f, a2 = 0.f;
#pragma unroll
  for (int j = 0; j < NTOK; ++j) {
    const unsigned short* vr = Vb + ((size_t)(b*NTOK + j)) * DMODEL + h * DHEAD;
    float w = sc[j] * inv;
    a0 += w*bf2f(vr[lane]); a1 += w*bf2f(vr[lane+32]); a2 += w*bf2f(vr[lane+64]);
  }
  unsigned short* o = pooled + (size_t)b * DMODEL + h * DHEAD;
  o[lane] = f2bf(a0); o[lane + 32] = f2bf(a1); o[lane + 64] = f2bf(a2);
}

// ---------- final logits dot (wave per row) ----------
__global__ void k_logits(const float* __restrict__ user_, const float* __restrict__ target_,
                         const float* __restrict__ fused_, const float* __restrict__ pooled_,
                         const float* __restrict__ qpool_, const float* __restrict__ lw,
                         const float* __restrict__ lb, float* __restrict__ out) {
  int lane = threadIdx.x & 31;
  int b = (int)((blockIdx.x * blockDim.x + threadIdx.x) >> 5);
  if (b >= BATCH) return;
  float acc = 0.f;
  for (int i = lane; i < 256; i += 32) acc += user_[(size_t)b*256 + i]  * lw[i];
  for (int i = lane; i < 704; i += 32) acc += target_[(size_t)b*704 + i]* lw[256 + i];
  for (int i = lane; i < 64;  i += 32) acc += fused_[(size_t)b*64 + i]  * lw[960 + i];
  for (int i = lane; i < 768; i += 32) acc += pooled_[(size_t)b*768 + i]* lw[1024 + i];
  for (int i = lane; i < 384; i += 32) acc += qpool_[(size_t)b*384 + i] * lw[1792 + i];
  acc = wave_sum(acc);
  if (lane == 0) out[b] = acc + lb[0];
}

// ---------- host ----------
static inline int wblocks(long waves) { return (int)((waves + 7) / 8); }

extern "C" void kernel_launch(void* const* d_in, const int* in_sizes, int n_in,
                              void* d_out, int out_size, void* d_ws, size_t ws_size,
                              hipStream_t stream) {
  (void)in_sizes; (void)n_in; (void)out_size; (void)ws_size;
  const float* user_emb   = (const float*)d_in[0];
  const float* ad_emb     = (const float*)d_in[1];
  const float* loc_emb    = (const float*)d_in[2];
  const float* cat_emb    = (const float*)d_in[3];
  const float* sq_emb     = (const float*)d_in[4];
  const float* sp_emb     = (const float*)d_in[5];
  const float* at_emb     = (const float*)d_in[6];
  const float* ap_emb     = (const float*)d_in[7];
  const float* click_emb  = (const float*)d_in[8];
  const float* ad_proj_w  = (const float*)d_in[9];
  const float* ad_proj_b  = (const float*)d_in[10];
  const float* wq         = (const float*)d_in[11];
  const float* wk         = (const float*)d_in[12];
  const float* wv         = (const float*)d_in[13];
  const float* fusion_w   = (const float*)d_in[14];
  const float* fusion_b   = (const float*)d_in[15];
  const float* agsom      = (const float*)d_in[16];
  const float* tpq_w      = (const float*)d_in[17];
  const float* tpq_b      = (const float*)d_in[18];
  const float* mha_q_w    = (const float*)d_in[19];
  const float* mha_k_w    = (const float*)d_in[20];
  const float* mha_v_w    = (const float*)d_in[21];
  const float* mha_o_w    = (const float*)d_in[22];
  const float* mha_o_b    = (const float*)d_in[23];
  const float* logits_w   = (const float*)d_in[24];
  const float* logits_b   = (const float*)d_in[25];
  const int*   features   = (const int*)d_in[26];
  float* out = (float*)d_out;

  char* ws = (char*)d_ws;
  size_t off = 0;
  auto alloc = [&](size_t bytes) -> void* {
    off = (off + 255) & ~(size_t)255;
    void* p = ws + off;
    off += bytes;
    return p;
  };
  const size_t NT = (size_t)BATCH * NTOK;        // 51200 tokens
  unsigned short* tok_b   = (unsigned short*)alloc(NT * DMODEL * 2);
  unsigned short* km_b    = (unsigned short*)alloc(NT * DMODEL * 2);
  unsigned short* vm_b    = (unsigned short*)alloc(NT * DMODEL * 2);
  int*            mask    = (int*)           alloc(NT * 4);
  float*          targ_f  = (float*)         alloc((size_t)BATCH * 704 * 4);
  unsigned short* targ_b  = (unsigned short*)alloc((size_t)BATCH * 704 * 2);
  float*          user_f  = (float*)         alloc((size_t)BATCH * 256 * 4);
  float*          qpool_f = (float*)         alloc((size_t)BATCH * 384 * 4);
  float*          item_f  = (float*)         alloc((size_t)BATCH * 64 * 4);
  unsigned short* fin_b   = (unsigned short*)alloc((size_t)BATCH * 256 * 2);
  float*          fused_f = (float*)         alloc((size_t)BATCH * 64 * 4);
  unsigned short* qt_b    = (unsigned short*)alloc((size_t)BATCH * DMODEL * 2);
  float*          q_f     = (float*)         alloc((size_t)BATCH * DMODEL * 4);
  unsigned short* pp_b    = (unsigned short*)alloc((size_t)BATCH * DMODEL * 2);
  float*          po_f    = (float*)         alloc((size_t)BATCH * DMODEL * 4);
  unsigned short* adp_t   = (unsigned short*)alloc((size_t)64 * 704 * 2);
  unsigned short* fus_t   = (unsigned short*)alloc((size_t)64 * 256 * 2);
  unsigned short* tpq_t   = (unsigned short*)alloc((size_t)768 * 704 * 2);
  unsigned short* qw_t    = (unsigned short*)alloc((size_t)768 * 768 * 2);
  unsigned short* kw_t    = (unsigned short*)alloc((size_t)768 * 768 * 2);
  unsigned short* vw_t    = (unsigned short*)alloc((size_t)768 * 768 * 2);
  unsigned short* ow_t    = (unsigned short*)alloc((size_t)768 * 768 * 2);

  auto XPOSE = [&](const float* W, unsigned short* Bt, int K, int N) {
    int n = K * N;
    k_transpose_bf16<<<(n + 255) / 256, 256, 0, stream>>>(W, Bt, K, N);
  };
  XPOSE(ad_proj_w, adp_t, 704, 64);
  XPOSE(fusion_w,  fus_t, 256, 64);
  XPOSE(tpq_w,     tpq_t, 704, 768);
  XPOSE(mha_q_w,   qw_t,  768, 768);
  XPOSE(mha_k_w,   kw_t,  768, 768);
  XPOSE(mha_v_w,   vw_t,  768, 768);
  XPOSE(mha_o_w,   ow_t,  768, 768);

  k_tokens<<<wblocks(NT), 256, 0, stream>>>(ad_emb, loc_emb, cat_emb, sq_emb, sp_emb,
                                            at_emb, ap_emb, click_emb, features, tok_b, mask);
  k_target_user<<<wblocks(BATCH), 256, 0, stream>>>(user_emb, ad_emb, loc_emb, cat_emb, sq_emb,
                                                    sp_emb, at_emb, ap_emb, features,
                                                    targ_f, targ_b, user_f);
  k_qpool<<<wblocks(BATCH), 256, 0, stream>>>(ad_emb, loc_emb, cat_emb, sq_emb, sp_emb,
                                              click_emb, features, qpool_f);

  auto GEMM = [&](const unsigned short* A, const unsigned short* Bt, const float* bias,
                  float* Cf, unsigned short* Cb, int M, int N, int K, int relu) {
    long tiles = (long)(M / 32) * (N / 32);
    k_gemm_wmma<<<wblocks(tiles), 256, 0, stream>>>(A, Bt, bias, Cf, Cb, M, N, K, relu);
  };

  // item = target @ ad_proj_w + b
  GEMM(targ_b, adp_t, ad_proj_b, item_f, nullptr, BATCH, 64, 704, 0);
  // AGSOM codebook + node attention -> fusion input (bf16)
  k_agsom<<<wblocks(BATCH), 256, 0, stream>>>(item_f, agsom, wq, wk, wv, fin_b);
  // fused = relu(fin @ fusion_w + b)
  GEMM(fin_b, fus_t, fusion_b, fused_f, nullptr, BATCH, 64, 256, 1);
  // qt = relu(target @ tpq_w + b)  (bf16 out, feeds Q proj)
  GEMM(targ_b, tpq_t, tpq_b, nullptr, qt_b, BATCH, DMODEL, 704, 1);
  // Q = qt @ mha_q_w  (f32 out, feeds attention)
  GEMM(qt_b, qw_t, nullptr, q_f, nullptr, BATCH, DMODEL, DMODEL, 0);
  // K, V projections over all 51200 tokens (dominant GEMMs, bf16 out)
  GEMM(tok_b, kw_t, nullptr, nullptr, km_b, (int)NT, DMODEL, DMODEL, 0);
  GEMM(tok_b, vw_t, nullptr, nullptr, vm_b, (int)NT, DMODEL, DMODEL, 0);
  // softmax pooling
  k_attn<<<wblocks((long)BATCH * NHEAD), 256, 0, stream>>>(q_f, km_b, vm_b, mask, pp_b);
  // pooled = attn_out @ mha_o_w + b
  GEMM(pp_b, ow_t, mha_o_b, po_f, nullptr, BATCH, DMODEL, DMODEL, 0);
  // final logits
  k_logits<<<wblocks(BATCH), 256, 0, stream>>>(user_f, targ_f, fused_f, po_f, qpool_f,
                                               logits_w, logits_b, out);
}
